// SparseDiffAttn_38379827757164
// MI455X (gfx1250) — matrix operand order repaired and checked by reading, not compiled, not run on hardware
//
#include <hip/hip_runtime.h>
#include <stdint.h>

// ---------------------------------------------------------------------------
// Grouped block-sparse attention for MI455X (gfx1250), wave32 + WMMA bf16.
//   q:[B,H,Q,D] f32, k/v:[B,H,KV,D] f32, inds:[B,H,G,KP] i32 -> out:[B,H,Q,D] f32
// One 512-thread block (16 waves) per (g,h,b); each wave owns 16 query rows.
// Flash-attention (exp2 domain) over KP in chunks of 64 gathered KV columns.
// Gather for chunk n+1 is issued AFTER the staging barrier so __syncthreads'
// implicit waitcnt-0 does not serialize it; the loads overlap chunk n's WMMAs.
// ---------------------------------------------------------------------------

typedef __attribute__((ext_vector_type(16))) __bf16 v16bf;
typedef __attribute__((ext_vector_type(8)))  float  v8f;

union BF16Frag {
    v16bf v;
    unsigned short u[16];
    uint4 q[2];
};

// native f32 -> bf16 (backend emits v_cvt_pk_bf16_f32)
__device__ __forceinline__ unsigned short f2bf(float f) {
    union { __bf16 h; unsigned short u; } c;
    c.h = (__bf16)f;
    return c.u;
}
__device__ __forceinline__ unsigned pkbf(float lo, float hi) {
    return (unsigned)f2bf(lo) | ((unsigned)f2bf(hi) << 16);
}
__device__ __forceinline__ float bf2f(unsigned short h) {
    union { unsigned u; float f; } c; c.u = ((unsigned)h) << 16;
    return c.f;
}

constexpr int B_  = 2;
constexpr int H_  = 16;
constexpr int Qt  = 4096;
constexpr int D_  = 128;
constexpr int KV_ = 10521;
constexpr int BM_ = 256;
constexpr int G_  = 16;
constexpr int KP_ = 1536;
constexpr int KN  = 64;            // KV columns per chunk
constexpr int NCH = KP_ / KN;      // 24 chunks
constexpr int NW  = 16;            // waves per block

__global__ __launch_bounds__(512, 1)
void sparse_attn_wmma(const float* __restrict__ q,
                      const float* __restrict__ k,
                      const float* __restrict__ v,
                      const int*   __restrict__ inds,
                      float* __restrict__ out)
{
    __shared__ __align__(16) unsigned short sK [KN * D_];      // [kvrow][d]  16 KB
    __shared__ __align__(16) unsigned short sVt[D_ * KN];      // [d][kvrow]  16 KB
    __shared__ __align__(16) unsigned short sP [NW * 16 * KN]; // per-wave P  32 KB

    const int g = blockIdx.x, h = blockIdx.y, b = blockIdx.z;
    const int tid  = threadIdx.x;
    const int lane = tid & 31;
    const int wave = tid >> 5;
    const int ln   = lane & 15;        // column lane (N) / row lane (M)
    const int hi   = lane >> 4;        // half-wave select
    const int rowbase = wave * 16;     // 16 query rows per wave

    const size_t bh = (size_t)b * H_ + h;
    const float* qg   = q + (bh * Qt + (size_t)g * BM_) * D_;
    const float* kbse = k + bh * (size_t)KV_ * D_;
    const float* vbse = v + bh * (size_t)KV_ * D_;
    const int*   ib   = inds + (bh * G_ + g) * (size_t)KP_;
    float*       outg = out + (bh * Qt + (size_t)g * BM_) * D_;

    // 1/sqrt(128) * log2(e): scores come out in log2 domain -> exp2 softmax
    const float scale = 0.08838834764831845f * 1.4426950408889634f;

    // ---- Q fragment (A layout, pre-scaled, bf16) held in registers --------
    BF16Frag qa[4];
    {
        const float* qrow = qg + (size_t)(rowbase + ln) * D_;
        #pragma unroll
        for (int c = 0; c < 4; ++c) {
            #pragma unroll
            for (int vv = 0; vv < 8; ++vv) {
                int k0 = c * 32 + ((vv < 4) ? 0 : 16) + (vv & 3) * 2 + hi * 8;
                qa[c].u[2 * vv]     = f2bf(qrow[k0]     * scale);
                qa[c].u[2 * vv + 1] = f2bf(qrow[k0 + 1] * scale);
            }
        }
    }

    // ---- flash state (C layout: lane ln = column, row = r + 8*hi) ---------
    v8f   acc[8];
    float mrow[8], lrow[8];
    #pragma unroll
    for (int dt = 0; dt < 8; ++dt) { v8f z = {}; acc[dt] = z; }
    #pragma unroll
    for (int r = 0; r < 8; ++r) { mrow[r] = -1e30f; lrow[r] = 0.0f; }

    const int grow = tid >> 3;   // gather: 8 threads per KV row
    const int gsub = tid & 7;

    // ---- software-pipelined gather: regs hold the *next* chunk's rows -----
    float4 kreg[4], vreg[4];
    auto issue_gather = [&](int ch) {
        int idx = ib[ch * KN + grow];
        idx = idx < 0 ? 0 : (idx >= KV_ ? KV_ - 1 : idx);
        const float4* krow = (const float4*)(kbse + (size_t)idx * D_);
        const float4* vrow = (const float4*)(vbse + (size_t)idx * D_);
        #pragma unroll
        for (int i = 0; i < 4; ++i) {
            kreg[i] = krow[gsub * 4 + i];
            vreg[i] = vrow[gsub * 4 + i];
        }
    };
    issue_gather(0);

    for (int ch = 0; ch < NCH; ++ch) {
        // ---- stage current chunk (regs -> bf16 LDS) ----------------------
        #pragma unroll
        for (int i = 0; i < 4; ++i) {
            int c4 = gsub * 4 + i;
            int d0 = c4 * 4;
            uint2 kp;
            kp.x = pkbf(kreg[i].x, kreg[i].y);
            kp.y = pkbf(kreg[i].z, kreg[i].w);
            *(uint2*)&sK[grow * D_ + d0] = kp;           // one b64 store
            sVt[(d0 + 0) * KN + grow] = f2bf(vreg[i].x); // V transposed
            sVt[(d0 + 1) * KN + grow] = f2bf(vreg[i].y);
            sVt[(d0 + 2) * KN + grow] = f2bf(vreg[i].z);
            sVt[(d0 + 3) * KN + grow] = f2bf(vreg[i].w);
        }
        __syncthreads();

        // Kick off next chunk's gather AFTER the barrier so the barrier's
        // implicit waitcnt-0 can't serialize it; the loads complete at the
        // end-of-compute barrier below, fully hidden behind the WMMAs.
        if (ch + 1 < NCH) issue_gather(ch + 1);

        unsigned short* sPw = sP + wave * (16 * KN);

        // ---- pass 1: S = Q K^T (log2 domain), row max, stash raw S -------
        float cmax[8];
        #pragma unroll
        for (int r = 0; r < 8; ++r) cmax[r] = -1e30f;

        #pragma unroll
        for (int j = 0; j < 4; ++j) {
            BF16Frag kf[4];
            const unsigned short* base = &sK[(j * 16 + ln) * D_ + hi * 8];
            #pragma unroll
            for (int c = 0; c < 4; ++c) {
                kf[c].q[0] = *(const uint4*)(base + c * 32);
                kf[c].q[1] = *(const uint4*)(base + c * 32 + 16);
            }
            v8f S = {};
            #pragma unroll
            for (int c = 0; c < 4; ++c)
                S = __builtin_amdgcn_wmma_f32_16x16x32_bf16(
                        false, qa[c].v, false, kf[c].v, (short)0, S,
                        false, false);
            #pragma unroll
            for (int r = 0; r < 8; ++r) {
                float s = S[r];
                cmax[r] = fmaxf(cmax[r], s);
                sPw[(r + 8 * hi) * KN + j * 16 + ln] = f2bf(s);
            }
        }

        // ---- row-max reduction (stays inside each 16-lane half) ----------
        float rescale[8];
        #pragma unroll
        for (int r = 0; r < 8; ++r) {
            float mv = cmax[r];
            mv = fmaxf(mv, __shfl_xor(mv, 1, 32));
            mv = fmaxf(mv, __shfl_xor(mv, 2, 32));
            mv = fmaxf(mv, __shfl_xor(mv, 4, 32));
            mv = fmaxf(mv, __shfl_xor(mv, 8, 32));
            float mn = fmaxf(mrow[r], mv);
            float sc = __builtin_amdgcn_exp2f(mrow[r] - mn);
            mrow[r] = mn;
            lrow[r] *= sc;
            rescale[r] = sc;
        }
        #pragma unroll
        for (int dt = 0; dt < 8; ++dt)
            #pragma unroll
            for (int r = 0; r < 8; ++r)
                acc[dt][r] *= rescale[r];

        // ---- broadcast per-row max into A layout (lane ln == row) --------
        float mA;
        {
            int srcl = (ln & 8) ? 16 : 0;
            int rsel = ln & 7;
            float got = 0.0f;
            #pragma unroll
            for (int r = 0; r < 8; ++r) {
                float tv = __shfl(mrow[r], srcl, 32);
                got = (rsel == r) ? tv : got;
            }
            mA = got;
        }

        // ---- pass 2: P = exp2(S - m) during A reload; O += P * V ---------
        float psum = 0.0f;
        #pragma unroll
        for (int kc = 0; kc < 2; ++kc) {
            BF16Frag pa;
            const unsigned short* src = &sPw[ln * KN + kc * 32 + hi * 8];
            pa.q[0] = *(const uint4*)(src);
            pa.q[1] = *(const uint4*)(src + 16);
            #pragma unroll
            for (int e = 0; e < 16; ++e) {
                float p = __builtin_amdgcn_exp2f(bf2f(pa.u[e]) - mA);
                psum += p;
                pa.u[e] = f2bf(p);
            }
            #pragma unroll
            for (int dt = 0; dt < 8; ++dt) {
                BF16Frag vf;
                const unsigned short* vs =
                    &sVt[(dt * 16 + ln) * KN + kc * 32 + hi * 8];
                vf.q[0] = *(const uint4*)(vs);
                vf.q[1] = *(const uint4*)(vs + 16);
                acc[dt] = __builtin_amdgcn_wmma_f32_16x16x32_bf16(
                    false, pa.v, false, vf.v, (short)0, acc[dt],
                    false, false);
            }
        }

        // ---- fold row sums back into C-layout running l ------------------
        psum += __shfl_xor(psum, 16, 32);
        #pragma unroll
        for (int r = 0; r < 8; ++r)
            lrow[r] += __shfl(psum, r + 8 * hi, 32);

        __syncthreads();   // compute done; prefetch loads also land here
    }

    // ---- epilogue: out = acc / l ------------------------------------------
    #pragma unroll
    for (int r = 0; r < 8; ++r) {
        int row = rowbase + r + 8 * hi;
        float inv = 1.0f / lrow[r];
        float* orow = outg + (size_t)row * D_;
        #pragma unroll
        for (int dt = 0; dt < 8; ++dt)
            orow[dt * 16 + ln] = acc[dt][r] * inv;
    }
}

extern "C" void kernel_launch(void* const* d_in, const int* in_sizes, int n_in,
                              void* d_out, int out_size, void* d_ws, size_t ws_size,
                              hipStream_t stream) {
    (void)in_sizes; (void)n_in; (void)out_size; (void)d_ws; (void)ws_size;
    const float* q    = (const float*)d_in[0];
    const float* k    = (const float*)d_in[1];
    const float* v    = (const float*)d_in[2];
    const int*   inds = (const int*)d_in[3];
    float* out = (float*)d_out;

    dim3 grid(G_, H_, B_);      // (16, 16, 2) = 512 workgroups
    sparse_attn_wmma<<<grid, 512, 0, stream>>>(q, k, v, inds, out);
}
